// Attention_Block_6433861010074
// MI455X (gfx1250) — matrix operand order repaired and checked by reading, not compiled, not run on hardware
//
#include <hip/hip_runtime.h>
#include <math.h>

// ---------------------------------------------------------------------------
// CDNA5 / gfx1250 transformer block. bf16 WMMA (16x16x32, f32 accumulate) for
// every matmul; fp32 VALU for LN / softmax / GELU; flash-attention single
// pass + deterministic mean-probs recompute pass. Register-tiled GEMMs
// (64x32 per wave) to keep L2 fragment traffic under the WMMA roofline.
// ---------------------------------------------------------------------------

typedef __attribute__((ext_vector_type(16))) __bf16 v16bf;
typedef __attribute__((ext_vector_type(8)))  __bf16 v8bf;
typedef __attribute__((ext_vector_type(8)))  float  v8f;

#define DIMC 1024
#define SEQ  2048
#define NB   2
#define NH   16
#define HD   64
#define NT   (NB * SEQ)      // 4096 total rows

static __device__ __forceinline__ __bf16 f2bf(float f) {
    unsigned int u = __builtin_bit_cast(unsigned int, f);
    unsigned int r = (u + 0x7fffu + ((u >> 16) & 1u)) >> 16;
    return __builtin_bit_cast(__bf16, (unsigned short)r);
}

static __device__ __forceinline__ v8f wmma_bf16(v16bf a, v16bf b, v8f c) {
    return __builtin_amdgcn_wmma_f32_16x16x32_bf16(
        false, a, false, b, (short)0, c, false, false);
}

// A fragment: 16x32 bf16, src pre-offset to (row0, k0), row-major, stride ld.
// Lanes 0-15: M=lane, K={0..7,16..23}; lanes 16-31: M=lane-16, K={8..15,24..31}.
static __device__ __forceinline__ v16bf load_frag_a(const __bf16* src, int ld, int lane) {
    int m = lane & 15, hi = lane >> 4;
    const __bf16* p = src + (size_t)m * ld + hi * 8;
    v8bf lo = *(const v8bf*)p;
    v8bf hh = *(const v8bf*)(p + 16);
    v16bf a;
#pragma unroll
    for (int i = 0; i < 8; i++) { a[i] = lo[i]; a[8 + i] = hh[i]; }
    return a;
}

// B fragment: 32x16 bf16, K contiguous per column. src pre-offset to
// (col0, k0); column n at src + n*ld. Lanes 0-15: N=lane, K={0..7,16..23};
// lanes 16-31: N=lane-16, K={8..15,24..31}.
static __device__ __forceinline__ v16bf load_frag_b(const __bf16* src, int ld, int lane) {
    int n = lane & 15, hi = lane >> 4;
    const __bf16* p = src + (size_t)n * ld + hi * 8;
    v8bf b0 = *(const v8bf*)p;
    v8bf b1 = *(const v8bf*)(p + 16);
    v16bf b;
#pragma unroll
    for (int i = 0; i < 8; i++) { b[i] = b0[i]; b[8 + i] = b1[i]; }
    return b;
}

// Row reductions across the 16 lanes sharing a C-matrix row (xor masks < 16
// never cross the half-wave boundary).
static __device__ __forceinline__ float redmax16(float v) {
    v = fmaxf(v, __shfl_xor(v, 1));
    v = fmaxf(v, __shfl_xor(v, 2));
    v = fmaxf(v, __shfl_xor(v, 4));
    v = fmaxf(v, __shfl_xor(v, 8));
    return v;
}
static __device__ __forceinline__ float redsum16(float v) {
    v += __shfl_xor(v, 1);
    v += __shfl_xor(v, 2);
    v += __shfl_xor(v, 4);
    v += __shfl_xor(v, 8);
    return v;
}

// ---------------------------------------------------------------------------
// Weight transpose + cast: W[k][n] fp32 -> Wt[n][k] bf16 (K contiguous).
// ---------------------------------------------------------------------------
__global__ __launch_bounds__(256) void wtrans_kernel(const float* __restrict__ W,
                                                     __bf16* __restrict__ Wt) {
    int idx = blockIdx.x * 256 + threadIdx.x;      // 1024*1024 elements
    int k = idx >> 10, n = idx & 1023;
    Wt[(size_t)n * DIMC + k] = f2bf(W[idx]);
}

// ---------------------------------------------------------------------------
// LayerNorm: fp32 in -> bf16 out, one block per row.
// ---------------------------------------------------------------------------
__global__ __launch_bounds__(256) void ln_kernel(const float* __restrict__ x,
                                                 const float* __restrict__ g,
                                                 const float* __restrict__ b,
                                                 __bf16* __restrict__ out) {
    __shared__ float red[256];
    int tid = threadIdx.x;
    size_t base = (size_t)blockIdx.x * DIMC;
    float v[4];
    float s = 0.f;
#pragma unroll
    for (int j = 0; j < 4; j++) { v[j] = x[base + tid + j * 256]; s += v[j]; }
    red[tid] = s; __syncthreads();
    for (int o = 128; o > 0; o >>= 1) { if (tid < o) red[tid] += red[tid + o]; __syncthreads(); }
    float mu = red[0] * (1.f / DIMC);
    __syncthreads();
    s = 0.f;
#pragma unroll
    for (int j = 0; j < 4; j++) { float d = v[j] - mu; s += d * d; }
    red[tid] = s; __syncthreads();
    for (int o = 128; o > 0; o >>= 1) { if (tid < o) red[tid] += red[tid + o]; __syncthreads(); }
    float rstd = rsqrtf(red[0] * (1.f / DIMC) + 1e-5f);
#pragma unroll
    for (int j = 0; j < 4; j++) {
        int i = tid + j * 256;
        out[base + i] = f2bf((v[j] - mu) * rstd * g[i] + b[i]);
    }
}

// ---------------------------------------------------------------------------
// QKV GEMM: A(ln1, NTx1024 bf16) x {WqT,WkT,WvT} -> q[b][h][s][d],
// k[b][h][s][d], vT[b][h][d][s] (all bf16, bias fused).
// Wave tile: 32 rows x 16 cols x 3 weight matrices (6 accumulators); A
// fragments reused across all three B operands.
// ---------------------------------------------------------------------------
__global__ __launch_bounds__(256) void qkv_gemm_kernel(
    const __bf16* __restrict__ A,
    const __bf16* __restrict__ WqT, const __bf16* __restrict__ WkT,
    const __bf16* __restrict__ WvT,
    const float* __restrict__ bq, const float* __restrict__ bk,
    const float* __restrict__ bv,
    __bf16* __restrict__ qo, __bf16* __restrict__ ko, __bf16* __restrict__ vto) {
    int lane = threadIdx.x & 31;
    int w = blockIdx.x * 8 + (threadIdx.x >> 5);
    int nt = w & 63, mg = w >> 6;                  // mg: 0..127 (2 m-tiles each)
    int m0 = mg * 32;
    v8f aq[2], ak[2], av[2];
    v8f z = {};
#pragma unroll
    for (int i = 0; i < 2; i++) { aq[i] = z; ak[i] = z; av[i] = z; }
    const __bf16* Abase = A + (size_t)m0 * DIMC;
    const __bf16* Bq = WqT + (size_t)nt * 16 * DIMC;
    const __bf16* Bk = WkT + (size_t)nt * 16 * DIMC;
    const __bf16* Bv = WvT + (size_t)nt * 16 * DIMC;
    for (int k0 = 0; k0 < DIMC; k0 += 32) {
        v16bf af[2];
#pragma unroll
        for (int i = 0; i < 2; i++)
            af[i] = load_frag_a(Abase + (size_t)i * 16 * DIMC + k0, DIMC, lane);
        v16bf fq = load_frag_b(Bq + k0, DIMC, lane);
        v16bf fk = load_frag_b(Bk + k0, DIMC, lane);
        v16bf fv = load_frag_b(Bv + k0, DIMC, lane);
#pragma unroll
        for (int i = 0; i < 2; i++) {
            aq[i] = wmma_bf16(af[i], fq, aq[i]);
            ak[i] = wmma_bf16(af[i], fk, ak[i]);
            av[i] = wmma_bf16(af[i], fv, av[i]);
        }
    }
    int hi = lane >> 4, ln = lane & 15;
    int n = nt * 16 + ln;
    int h = n >> 6, d = n & 63;
    float cq = bq[n], ck = bk[n], cv = bv[n];
#pragma unroll
    for (int i = 0; i < 2; i++)
#pragma unroll
        for (int r = 0; r < 8; r++) {
            int rowg = m0 + i * 16 + r + hi * 8;
            int bb = rowg >> 11, s = rowg & 2047;
            size_t hbase = (size_t)(bb * NH + h);
            qo[(hbase * SEQ + s) * HD + d] = f2bf(aq[i][r] + cq);
            ko[(hbase * SEQ + s) * HD + d] = f2bf(ak[i][r] + ck);
            vto[(hbase * HD + d) * SEQ + s] = f2bf(av[i][r] + cv);
        }
}

// ---------------------------------------------------------------------------
// Generic 1024x1024 GEMM with fused epilogue. Wave tile: 64x32 (4 M-tiles x
// 2 N-tiles, 8 accumulators) -> A traffic /2, B traffic /4 vs 16x16 tiles.
// mode 0: out_f32 = resid + acc + bias   (O-proj / MLP2 residual adds)
// mode 1: out_bf16 = gelu(acc + bias)    (MLP1)
// ---------------------------------------------------------------------------
__global__ __launch_bounds__(256) void gemm_fused_kernel(
    const __bf16* __restrict__ A, const __bf16* __restrict__ Wt,
    const float* __restrict__ bias, const float* __restrict__ resid,
    float* __restrict__ outf, __bf16* __restrict__ outbf, int mode) {
    int lane = threadIdx.x & 31;
    int w = blockIdx.x * 8 + (threadIdx.x >> 5);
    int ng = w & 31, mg = w >> 5;                  // ng: 0..31, mg: 0..63
    int m0 = mg * 64, n0 = ng * 32;
    v8f acc[4][2];
    v8f z = {};
#pragma unroll
    for (int i = 0; i < 4; i++)
#pragma unroll
        for (int j = 0; j < 2; j++) acc[i][j] = z;
    const __bf16* Abase = A + (size_t)m0 * DIMC;
    for (int k0 = 0; k0 < DIMC; k0 += 32) {
        v16bf af[4];
#pragma unroll
        for (int i = 0; i < 4; i++)
            af[i] = load_frag_a(Abase + (size_t)i * 16 * DIMC + k0, DIMC, lane);
        v16bf bf_[2];
#pragma unroll
        for (int j = 0; j < 2; j++)
            bf_[j] = load_frag_b(Wt + (size_t)(n0 + j * 16) * DIMC + k0, DIMC, lane);
#pragma unroll
        for (int i = 0; i < 4; i++)
#pragma unroll
            for (int j = 0; j < 2; j++)
                acc[i][j] = wmma_bf16(af[i], bf_[j], acc[i][j]);
    }
    int hi = lane >> 4, ln = lane & 15;
#pragma unroll
    for (int j = 0; j < 2; j++) {
        int n = n0 + j * 16 + ln;
        float bc = bias[n];
#pragma unroll
        for (int i = 0; i < 4; i++)
#pragma unroll
            for (int r = 0; r < 8; r++) {
                size_t idx = (size_t)(m0 + i * 16 + r + hi * 8) * DIMC + n;
                float t = acc[i][j][r] + bc;
                if (mode == 0) {
                    outf[idx] = resid[idx] + t;
                } else {
                    float gl = 0.5f * t * (1.0f + erff(t * 0.70710678118654752f));
                    outbf[idx] = f2bf(gl);
                }
            }
    }
}

// ---------------------------------------------------------------------------
// Flash attention: one wave per (b, h, 16-row q tile). Online softmax in
// registers, P routed through LDS (C-layout f32 -> A-layout bf16), ctx via
// WMMA. Writes ctx[b][s][h*64+d] bf16 and final (m, l) per row.
// ---------------------------------------------------------------------------
__global__ __launch_bounds__(256) void attn_flash_kernel(
    const __bf16* __restrict__ q, const __bf16* __restrict__ k,
    const __bf16* __restrict__ vt, __bf16* __restrict__ ctx,
    float* __restrict__ mbuf, float* __restrict__ lbuf) {
    __shared__ __align__(16) __bf16 pbuf[8][16 * 32];
    int lane = threadIdx.x & 31, wv = threadIdx.x >> 5;
    int w = blockIdx.x * 8 + wv;
    int qt = w & 127, h = (w >> 7) & 15, b = w >> 11;
    int q0 = qt * 16;
    int hi = lane >> 4, ln = lane & 15;

    const __bf16* qh = q  + (size_t)(b * NH + h) * SEQ * HD;
    const __bf16* kh = k  + (size_t)(b * NH + h) * SEQ * HD;
    const __bf16* vh = vt + (size_t)(b * NH + h) * HD * SEQ;

    v16bf qa0 = load_frag_a(qh + (size_t)q0 * HD + 0,  HD, lane);
    v16bf qa1 = load_frag_a(qh + (size_t)q0 * HD + 32, HD, lane);

    float mr[8], lr[8];
    v8f cacc[4];
    v8f z = {};
#pragma unroll
    for (int r = 0; r < 8; r++) { mr[r] = -INFINITY; lr[r] = 0.f; }
#pragma unroll
    for (int t = 0; t < 4; t++) cacc[t] = z;

    int nch = (q0 + 15) / 32 + 1;
    for (int c = 0; c < nch; c++) {
        int kb = c * 32;
        v8f sc[2];
#pragma unroll
        for (int t = 0; t < 2; t++) {
            const __bf16* kcol = kh + (size_t)(kb + t * 16) * HD;
            v8f s0 = {};
            s0 = wmma_bf16(qa0, load_frag_b(kcol + 0,  HD, lane), s0);
            s0 = wmma_bf16(qa1, load_frag_b(kcol + 32, HD, lane), s0);
            sc[t] = s0;
        }
        float cmax[8];
#pragma unroll
        for (int r = 0; r < 8; r++) cmax[r] = -INFINITY;
#pragma unroll
        for (int t = 0; t < 2; t++)
#pragma unroll
            for (int r = 0; r < 8; r++) {
                int key = kb + t * 16 + ln;
                int qr  = q0 + r + hi * 8;
                float s = sc[t][r] * 0.125f;
                if (key > qr) s = -INFINITY;
                sc[t][r] = s;
                cmax[r] = fmaxf(cmax[r], s);
            }
        float alpha[8], rs[8];
#pragma unroll
        for (int r = 0; r < 8; r++) {
            float cm = redmax16(cmax[r]);
            float mn = fmaxf(mr[r], cm);
            alpha[r] = __expf(mr[r] - mn);
            mr[r] = mn;
            rs[r] = 0.f;
        }
#pragma unroll
        for (int t = 0; t < 2; t++)
#pragma unroll
            for (int r = 0; r < 8; r++) {
                float p = __expf(sc[t][r] - mr[r]);
                sc[t][r] = p;
                rs[r] += p;
            }
#pragma unroll
        for (int r = 0; r < 8; r++) lr[r] = lr[r] * alpha[r] + redsum16(rs[r]);
#pragma unroll
        for (int t = 0; t < 4; t++)
#pragma unroll
            for (int r = 0; r < 8; r++) cacc[t][r] *= alpha[r];
        // P: f32 C layout -> bf16 A layout via LDS bounce
#pragma unroll
        for (int t = 0; t < 2; t++)
#pragma unroll
            for (int r = 0; r < 8; r++)
                pbuf[wv][(r + hi * 8) * 32 + t * 16 + ln] = f2bf(sc[t][r]);
        asm volatile("s_wait_dscnt 0" ::: "memory");
        v16bf pa = load_frag_a(&pbuf[wv][0], 32, lane);
#pragma unroll
        for (int dt = 0; dt < 4; dt++)
            cacc[dt] = wmma_bf16(pa,
                load_frag_b(vh + (size_t)(dt * 16) * SEQ + kb, SEQ, lane), cacc[dt]);
    }
    // epilogue: normalize + store ctx, store (m, l)
#pragma unroll
    for (int r = 0; r < 8; r++) {
        float inv = 1.0f / lr[r];
#pragma unroll
        for (int dt = 0; dt < 4; dt++) {
            size_t idx = (size_t)(b * SEQ + q0 + r + hi * 8) * DIMC
                       + h * HD + dt * 16 + ln;
            ctx[idx] = f2bf(cacc[dt][r] * inv);
        }
    }
    if (ln == 0) {
#pragma unroll
        for (int r = 0; r < 8; r++) {
            int row = q0 + r + hi * 8;
            mbuf[(b * NH + h) * SEQ + row] = mr[r];
            lbuf[(b * NH + h) * SEQ + row] = lr[r];
        }
    }
}

// ---------------------------------------------------------------------------
// Mean attention probs: one wave per (b, q-tile16, k-chunk32); loops the 16
// heads in registers (deterministic, no atomics), recomputing scores via WMMA
// and using saved (m, l). Writes a.mean(axis=1) -> (b, S, S) fp32.
// ---------------------------------------------------------------------------
__global__ __launch_bounds__(256) void attn_mean_kernel(
    const __bf16* __restrict__ q, const __bf16* __restrict__ k,
    const float* __restrict__ mbuf, const float* __restrict__ lbuf,
    float* __restrict__ amean) {
    int lane = threadIdx.x & 31;
    int w = blockIdx.x * 8 + (threadIdx.x >> 5);
    int kc = w & 63, qt = (w >> 6) & 127, b = w >> 13;
    int q0 = qt * 16, kb = kc * 32;
    int hi = lane >> 4, ln = lane & 15;
    float* ob = amean + (size_t)b * SEQ * SEQ;

    if (kb > q0 + 15) {  // fully masked tile -> zeros
#pragma unroll
        for (int t = 0; t < 2; t++)
#pragma unroll
            for (int r = 0; r < 8; r++)
                ob[(size_t)(q0 + r + hi * 8) * SEQ + kb + t * 16 + ln] = 0.f;
        return;
    }
    v8f acc[2];
    v8f z = {};
    acc[0] = z; acc[1] = z;
    for (int h = 0; h < NH; h++) {
        const __bf16* qh = q + (size_t)(b * NH + h) * SEQ * HD;
        const __bf16* kh = k + (size_t)(b * NH + h) * SEQ * HD;
        v16bf qa0 = load_frag_a(qh + (size_t)q0 * HD + 0,  HD, lane);
        v16bf qa1 = load_frag_a(qh + (size_t)q0 * HD + 32, HD, lane);
        float mv[8], il[8];
#pragma unroll
        for (int r = 0; r < 8; r++) {
            int row = (b * NH + h) * SEQ + q0 + r + hi * 8;
            mv[r] = mbuf[row];
            il[r] = 1.0f / lbuf[row];
        }
#pragma unroll
        for (int t = 0; t < 2; t++) {
            const __bf16* kcol = kh + (size_t)(kb + t * 16) * HD;
            v8f s0 = {};
            s0 = wmma_bf16(qa0, load_frag_b(kcol + 0,  HD, lane), s0);
            s0 = wmma_bf16(qa1, load_frag_b(kcol + 32, HD, lane), s0);
#pragma unroll
            for (int r = 0; r < 8; r++) {
                int key = kb + t * 16 + ln;
                int qr  = q0 + r + hi * 8;
                float p = (key > qr) ? 0.f
                        : __expf(s0[r] * 0.125f - mv[r]) * il[r];
                acc[t][r] += p;
            }
        }
    }
#pragma unroll
    for (int t = 0; t < 2; t++)
#pragma unroll
        for (int r = 0; r < 8; r++)
            ob[(size_t)(q0 + r + hi * 8) * SEQ + kb + t * 16 + ln] =
                acc[t][r] * (1.0f / NH);
}

// ---------------------------------------------------------------------------
// Host-side orchestration.
// ---------------------------------------------------------------------------
extern "C" void kernel_launch(void* const* d_in, const int* in_sizes, int n_in,
                              void* d_out, int out_size, void* d_ws, size_t ws_size,
                              hipStream_t stream) {
    (void)in_sizes; (void)n_in; (void)out_size; (void)ws_size;
    const float* x   = (const float*)d_in[0];
    const float* Wq  = (const float*)d_in[1];  const float* bq  = (const float*)d_in[2];
    const float* Wk  = (const float*)d_in[3];  const float* bk  = (const float*)d_in[4];
    const float* Wv  = (const float*)d_in[5];  const float* bv  = (const float*)d_in[6];
    const float* Wo  = (const float*)d_in[7];  const float* bo  = (const float*)d_in[8];
    const float* g1  = (const float*)d_in[9];  const float* b1  = (const float*)d_in[10];
    const float* g2  = (const float*)d_in[11]; const float* b2  = (const float*)d_in[12];
    const float* Wm1 = (const float*)d_in[13]; const float* bm1 = (const float*)d_in[14];
    const float* Wm2 = (const float*)d_in[15]; const float* bm2 = (const float*)d_in[16];

    float* out_x  = (float*)d_out;
    float* out_am = out_x + (size_t)NT * DIMC;

    // workspace carve-up
    char* p = (char*)d_ws;
    size_t off = 0;
    auto take = [&](size_t bytes) { char* r = p + off; off += (bytes + 255) & ~(size_t)255; return r; };
    __bf16* ln_buf = (__bf16*)take((size_t)NT * DIMC * 2);
    __bf16* WqT = (__bf16*)take((size_t)DIMC * DIMC * 2);
    __bf16* WkT = (__bf16*)take((size_t)DIMC * DIMC * 2);
    __bf16* WvT = (__bf16*)take((size_t)DIMC * DIMC * 2);
    __bf16* WoT = (__bf16*)take((size_t)DIMC * DIMC * 2);
    __bf16* Wm1T = (__bf16*)take((size_t)DIMC * DIMC * 2);
    __bf16* Wm2T = (__bf16*)take((size_t)DIMC * DIMC * 2);
    __bf16* qb  = (__bf16*)take((size_t)NT * DIMC * 2);
    __bf16* kb_ = (__bf16*)take((size_t)NT * DIMC * 2);
    __bf16* vtb = (__bf16*)take((size_t)NT * DIMC * 2);
    __bf16* ctxb = (__bf16*)take((size_t)NT * DIMC * 2);
    __bf16* hb  = (__bf16*)take((size_t)NT * DIMC * 2);
    float*  x2  = (float*)take((size_t)NT * DIMC * 4);
    float*  mbuf = (float*)take((size_t)NB * NH * SEQ * 4);
    float*  lbuf = (float*)take((size_t)NB * NH * SEQ * 4);

    const int WT_BLK = (DIMC * DIMC) / 256;           // 4096
    wtrans_kernel<<<WT_BLK, 256, 0, stream>>>(Wq,  WqT);
    wtrans_kernel<<<WT_BLK, 256, 0, stream>>>(Wk,  WkT);
    wtrans_kernel<<<WT_BLK, 256, 0, stream>>>(Wv,  WvT);
    wtrans_kernel<<<WT_BLK, 256, 0, stream>>>(Wo,  WoT);
    wtrans_kernel<<<WT_BLK, 256, 0, stream>>>(Wm1, Wm1T);
    wtrans_kernel<<<WT_BLK, 256, 0, stream>>>(Wm2, Wm2T);

    // LN1
    ln_kernel<<<NT, 256, 0, stream>>>(x, g1, b1, ln_buf);
    // QKV: 128 m-groups (32 rows) * 64 n-tiles -> 8192 waves / 8 per block
    qkv_gemm_kernel<<<1024, 256, 0, stream>>>(ln_buf, WqT, WkT, WvT,
                                              bq, bk, bv, qb, kb_, vtb);
    // flash attention: 2*16*128 waves / 8
    attn_flash_kernel<<<512, 256, 0, stream>>>(qb, kb_, vtb, ctxb, mbuf, lbuf);
    // mean probs: 2*128*64 waves / 8
    attn_mean_kernel<<<2048, 256, 0, stream>>>(qb, kb_, mbuf, lbuf, out_am);
    // O-proj + residual: x2 = x + ctx @ Wo + bo
    // 64 m-groups (64 rows) * 32 n-groups (32 cols) -> 2048 waves / 8
    gemm_fused_kernel<<<256, 256, 0, stream>>>(ctxb, WoT, bo, x, x2, nullptr, 0);
    // LN2
    ln_kernel<<<NT, 256, 0, stream>>>(x2, g2, b2, ln_buf);
    // MLP1 + GELU (exact)
    gemm_fused_kernel<<<256, 256, 0, stream>>>(ln_buf, Wm1T, bm1, nullptr, nullptr, hb, 1);
    // MLP2 + residual -> final x output
    gemm_fused_kernel<<<256, 256, 0, stream>>>(hb, Wm2T, bm2, x2, out_x, nullptr, 0);
}
